// HydraAttention_19404662243630
// MI455X (gfx1250) — matrix operand order repaired and checked by reading, not compiled, not run on hardware
//
#include <hip/hip_runtime.h>
#include <hip/hip_bf16.h>

// ---------------------------------------------------------------------------
// HydraAttention on gfx1250 (MI455X): bf16 WMMA GEMMs, LDS double-buffered
// tiles with transpose-at-write, 2x2 register blocking per wave (4 wmma per
// K-step per wave), plus fused linear-attention reductions.
// B=8, C=512, H=W=64 -> N=4096 tokens, 3C=1536.
// ---------------------------------------------------------------------------

typedef __attribute__((ext_vector_type(16))) __bf16 v16bf;
typedef __attribute__((ext_vector_type(8)))  __bf16 v8bf;
typedef __attribute__((ext_vector_type(8)))  float  v8f;

#define NB   8
#define CH   512
#define NT   4096           // tokens per batch (64*64)
#define C3   1536

#define LDS_PITCH 48        // bf16 elements per LDS row (96B, 32B-aligned)

// Build an A fragment (16x32 bf16) from a K-contiguous LDS row.
// Layout: a[0..7] = k = khalf*16 + 0..7 ; a[8..15] = k = khalf*16 + 8 + 0..7.
__device__ inline v16bf make_afrag(const __bf16* row, int khalf) {
  const v8bf lo = *reinterpret_cast<const v8bf*>(row + khalf * 16);
  const v8bf hi = *reinterpret_cast<const v8bf*>(row + khalf * 16 + 8);
  v16bf a;
#pragma unroll
  for (int e = 0; e < 8; ++e) { a[e] = lo[e]; a[8 + e] = hi[e]; }
  return a;
}

// ---------------- generic tiled GEMM: C[M][N] = A[M][K] * B[K][N] ----------
// A row-major (lda = K), B row-major (ldb = N), C row-major (ldc = N).
// Block = 256 threads (8 waves). Block tile 64M x 128N, K-step 32.
// Wave tile 32M x 32N = 2x2 wmma. LDS double-buffered, 1 barrier / K-step.

template <typename OutT, bool BIAS_ON_M>
__global__ __launch_bounds__(256)
void gemm_bf16_wmma(const __bf16* __restrict__ Ag, size_t aBatch,
                    const __bf16* __restrict__ Bg, size_t bBatch,
                    const float*  __restrict__ bias,
                    OutT* __restrict__ Cg, size_t cBatch,
                    int K, int ldn /* = N = ldb = ldc */) {
  __shared__ __bf16 As[2][64][LDS_PITCH];    // [buf][m-local][k]   12 KiB
  __shared__ __bf16 Bs[2][128][LDS_PITCH];   // [buf][n-local][k]   24 KiB

  const int tid  = threadIdx.x;
  const int lane = tid & 31;
  const int wave = tid >> 5;
  const int b    = blockIdx.z;
  const int n0   = blockIdx.x * 128;
  const int m0   = blockIdx.y * 64;

  const __bf16* A = Ag + (size_t)b * aBatch;
  const __bf16* B = Bg + (size_t)b * bBatch;
  OutT*         C = Cg + (size_t)b * cBatch;

  // staging roles:
  //  A tile 64x32: thread loads 8 K-contiguous bf16 (b128).
  const int ar = tid >> 2;            // 0..63  (m-local)
  const int ac = (tid & 3) * 8;       // 0..24  (k)
  //  B tile 32x128: two reps, each 8 N-contiguous bf16 (b128) scattered to
  //  Bs[n][k] (transpose at write).
  //  rep r: tid2 = tid + r*256 ; br = tid2>>4 (k) ; bn = (tid2&15)*8 (n-local)

  // wave tile coordinates: 2 waves along M (32 each), 4 along N (32 each)
  const int mw    = (wave >> 2) * 32;
  const int nw    = (wave & 3) * 32;
  const int mrow  = mw + (lane & 15);
  const int nrow  = nw + (lane & 15);
  const int khalf = lane >> 4;

  uint4 avec, bvec0, bvec1;

  auto load_tiles = [&](int kk) {
    avec  = *reinterpret_cast<const uint4*>(A + (size_t)(m0 + ar) * K + kk + ac);
    {
      const int t0 = tid,        br0 = t0 >> 4, bn0 = (t0 & 15) * 8;
      const int t1 = tid + 256,  br1 = t1 >> 4, bn1 = (t1 & 15) * 8;
      bvec0 = *reinterpret_cast<const uint4*>(B + (size_t)(kk + br0) * ldn + n0 + bn0);
      bvec1 = *reinterpret_cast<const uint4*>(B + (size_t)(kk + br1) * ldn + n0 + bn1);
    }
  };

  auto store_tiles = [&](int buf) {
    *reinterpret_cast<uint4*>(&As[buf][ar][ac]) = avec;
    const v8bf b0 = __builtin_bit_cast(v8bf, bvec0);
    const v8bf b1 = __builtin_bit_cast(v8bf, bvec1);
    const int t0 = tid,        br0 = t0 >> 4, bn0 = (t0 & 15) * 8;
    const int t1 = tid + 256,  br1 = t1 >> 4, bn1 = (t1 & 15) * 8;
#pragma unroll
    for (int e = 0; e < 8; ++e) Bs[buf][bn0 + e][br0] = b0[e];
#pragma unroll
    for (int e = 0; e < 8; ++e) Bs[buf][bn1 + e][br1] = b1[e];
  };

  v8f acc00 = {}, acc01 = {}, acc10 = {}, acc11 = {};

  // prologue: stage K-step 0 into buffer 0
  load_tiles(0);
  store_tiles(0);
  __syncthreads();

  int buf = 0;
  for (int kk = 0; kk < K; kk += 32) {
    const bool has_next = (kk + 32) < K;
    if (has_next) {
      load_tiles(kk + 32);   // global loads overlap the wmma burst below
      __builtin_prefetch(A + (size_t)(m0 + ar) * K + kk + 64 + ac, 0, 1);
      __builtin_prefetch(B + (size_t)(kk + 64 + (tid >> 4)) * ldn + n0 + (tid & 15) * 8, 0, 1);
    }

    // fragments from current buffer (all contiguous ds_load_b128)
    const v16bf af0 = make_afrag(&As[buf][mrow][0],      khalf);
    const v16bf af1 = make_afrag(&As[buf][mrow + 16][0], khalf);
    const v16bf bf0 = *reinterpret_cast<const v16bf*>(&Bs[buf][nrow][khalf * 16]);
    const v16bf bf1 = *reinterpret_cast<const v16bf*>(&Bs[buf][nrow + 16][khalf * 16]);

    acc00 = __builtin_amdgcn_wmma_f32_16x16x32_bf16(false, af0, false, bf0,
                                                    (short)0, acc00, false, false);
    acc01 = __builtin_amdgcn_wmma_f32_16x16x32_bf16(false, af0, false, bf1,
                                                    (short)0, acc01, false, false);
    acc10 = __builtin_amdgcn_wmma_f32_16x16x32_bf16(false, af1, false, bf0,
                                                    (short)0, acc10, false, false);
    acc11 = __builtin_amdgcn_wmma_f32_16x16x32_bf16(false, af1, false, bf1,
                                                    (short)0, acc11, false, false);

    if (has_next) store_tiles(buf ^ 1);   // other buffer: no WAR with readers
    __syncthreads();                      // publish for next iteration
    buf ^= 1;
  }

  // D layout: N = lane&15, M = v + 8*(lane>>4)
  const int mh = (lane >> 4) * 8;
  const int nn = lane & 15;
#pragma unroll
  for (int mi = 0; mi < 2; ++mi) {
    const v8f* accs[2] = { mi ? &acc10 : &acc00, mi ? &acc11 : &acc01 };
#pragma unroll
    for (int ni = 0; ni < 2; ++ni) {
      const v8f a = *accs[ni];
      const int gn = n0 + nw + ni * 16 + nn;
#pragma unroll
      for (int v = 0; v < 8; ++v) {
        const int gm = m0 + mw + mi * 16 + v + mh;
        const float val = a[v] + (BIAS_ON_M ? bias[gm] : bias[gn]);
        C[(size_t)gm * ldn + gn] = (OutT)val;
      }
    }
  }
}

// ---------------- precision-conversion kernels ----------------

__global__ void cvt_f32_to_bf16(const float* __restrict__ in,
                                __bf16* __restrict__ out, int n) {
  int i = blockIdx.x * 256 + threadIdx.x;
  if (i < n) out[i] = (__bf16)in[i];
}

// W_qkv (C x 3C, row-major) -> W_qkvT (3C x C, row-major) in bf16
__global__ void cvt_wqkv_transpose(const float* __restrict__ w,
                                   __bf16* __restrict__ wt) {
  int i = blockIdx.x * 256 + threadIdx.x;   // over 1536*512
  int j = i >> 9;
  int c = i & 511;
  wt[i] = (__bf16)w[c * C3 + j];
}

// ---------------- per-token inverse L2 norms of q and k ----------

__global__ void norms_kernel(const __bf16* __restrict__ qkvt,
                             float* __restrict__ inq,
                             float* __restrict__ ink) {
  const int b = blockIdx.y;
  const int t = blockIdx.x * 256 + threadIdx.x;
  const __bf16* base = qkvt + (size_t)b * C3 * NT + t;
  float sq = 0.f, sk = 0.f;
#pragma unroll 4
  for (int c = 0; c < CH; ++c) {
    float qv = (float)base[(size_t)c * NT];
    float kv = (float)base[(size_t)(CH + c) * NT];
    sq += qv * qv;
    sk += kv * kv;
  }
  inq[b * NT + t] = rsqrtf(sq);
  ink[b * NT + t] = rsqrtf(sk);
}

// ---------------- s_know/s_mask[b][c] = sum_t (1-m | m) * k_hat * v ---------

__global__ void sums_kernel(const __bf16* __restrict__ qkvt,
                            const float* __restrict__ ink,
                            const float* __restrict__ mask,
                            float* __restrict__ sknow,
                            float* __restrict__ smask) {
  const int c = blockIdx.x;
  const int b = blockIdx.y;
  const __bf16* kb = qkvt + ((size_t)b * C3 + CH + c)     * NT;
  const __bf16* vb = qkvt + ((size_t)b * C3 + 2 * CH + c) * NT;
  const float*  mb = mask + (size_t)b * NT;
  const float* ikb = ink  + (size_t)b * NT;

  float pk = 0.f, pm = 0.f;
  for (int t = threadIdx.x; t < NT; t += 256) {
    float kv = (float)kb[t] * ikb[t] * (float)vb[t];
    float m  = mb[t];
    pk += (1.0f - m) * kv;
    pm += m * kv;
  }
  __shared__ float r1[256], r2[256];
  r1[threadIdx.x] = pk;
  r2[threadIdx.x] = pm;
  __syncthreads();
  for (int s = 128; s > 0; s >>= 1) {
    if (threadIdx.x < s) {
      r1[threadIdx.x] += r1[threadIdx.x + s];
      r2[threadIdx.x] += r2[threadIdx.x + s];
    }
    __syncthreads();
  }
  if (threadIdx.x == 0) {
    sknow[b * CH + c] = r1[0];
    smask[b * CH + c] = r2[0];
  }
}

// ---------------- A2[b][t][c] = q_hat * coef  (LDS transpose) ---------------

__global__ void a2_kernel(const __bf16* __restrict__ qkvt,
                          const float* __restrict__ inq,
                          const float* __restrict__ mask,
                          const float* __restrict__ sknow,
                          const float* __restrict__ smask,
                          __bf16* __restrict__ a2) {
  __shared__ float tile[64][65];
  const int b  = blockIdx.z;
  const int c0 = blockIdx.x * 64;
  const int t0 = blockIdx.y * 64;
  const __bf16* qb = qkvt + (size_t)b * C3 * NT;

  for (int i = threadIdx.x; i < 64 * 64; i += 256) {
    const int cc = i >> 6, tt = i & 63;
    tile[cc][tt] = (float)qb[(size_t)(c0 + cc) * NT + t0 + tt];
  }
  __syncthreads();
  for (int i = threadIdx.x; i < 64 * 64; i += 256) {
    const int tt = i >> 6, cc = i & 63;
    const int t = t0 + tt, c = c0 + cc;
    const float m   = mask[b * NT + t];
    const float inv = 1.0f - m;
    const float sk  = sknow[b * CH + c];
    const float sm  = smask[b * CH + c];
    const float coef = sk * inv * inv + m * (0.3f * sk * inv + 0.7f * sm * m);
    const float qhat = tile[cc][tt] * inq[b * NT + t];
    a2[((size_t)b * NT + t) * CH + c] = (__bf16)(qhat * coef);
  }
}

// ---------------- host side ----------------

extern "C" void kernel_launch(void* const* d_in, const int* in_sizes, int n_in,
                              void* d_out, int out_size, void* d_ws, size_t ws_size,
                              hipStream_t stream) {
  const float* x     = (const float*)d_in[0];  // (8,512,64,64)
  const float* mask  = (const float*)d_in[1];  // (8,1,64,64)
  const float* Wqkv  = (const float*)d_in[2];  // (512,1536)
  const float* bqkv  = (const float*)d_in[3];  // (1536,)
  const float* Wout  = (const float*)d_in[4];  // (512,512)
  const float* bout  = (const float*)d_in[5];  // (512,)
  float* out = (float*)d_out;                  // (8,4096,512)

  // workspace carve-up (all sizes are multiples of 256 bytes)
  char* w = (char*)d_ws;
  __bf16* xbf    = (__bf16*)w; w += (size_t)NB * CH * NT * 2;   // 32 MiB
  __bf16* wqkvt  = (__bf16*)w; w += (size_t)C3 * CH * 2;        // 1.5 MiB
  __bf16* woutbf = (__bf16*)w; w += (size_t)CH * CH * 2;        // 0.5 MiB
  __bf16* qkvt   = (__bf16*)w; w += (size_t)NB * C3 * NT * 2;   // 96 MiB
  float*  inq    = (float*)w;  w += (size_t)NB * NT * 4;        // 128 KiB
  float*  ink    = (float*)w;  w += (size_t)NB * NT * 4;        // 128 KiB
  float*  sknow  = (float*)w;  w += (size_t)NB * CH * 4;        // 16 KiB
  float*  smask  = (float*)w;  w += (size_t)NB * CH * 4;        // 16 KiB
  __bf16* a2     = (__bf16*)w; w += (size_t)NB * NT * CH * 2;   // 32 MiB

  // 0) precision conversion
  {
    const int nX = NB * CH * NT;
    cvt_f32_to_bf16<<<dim3((nX + 255) / 256), dim3(256), 0, stream>>>(x, xbf, nX);
    const int nWq = C3 * CH;
    cvt_wqkv_transpose<<<dim3((nWq + 255) / 256), dim3(256), 0, stream>>>(Wqkv, wqkvt);
    const int nWo = CH * CH;
    cvt_f32_to_bf16<<<dim3((nWo + 255) / 256), dim3(256), 0, stream>>>(Wout, woutbf, nWo);
  }

  // 1) QKV GEMM, transposed orientation: qkvT[b][j][t] = WqkvT[j][:]·x[b][:][t]
  //    M = 3C (bias on M), N = NT, K = CH; bf16 output.
  gemm_bf16_wmma<__bf16, true>
      <<<dim3(NT / 128, C3 / 64, NB), dim3(256), 0, stream>>>(
          wqkvt, (size_t)0,
          xbf,   (size_t)CH * NT,
          bqkv,
          qkvt,  (size_t)C3 * NT,
          CH, NT);

  // 2a) per-token inverse norms
  norms_kernel<<<dim3(NT / 256, NB), dim3(256), 0, stream>>>(qkvt, inq, ink);

  // 2b) masked channel-wise sums of k_hat * v
  sums_kernel<<<dim3(CH, NB), dim3(256), 0, stream>>>(qkvt, ink, mask, sknow, smask);

  // 2c) fuse coefficients + transpose to (b,t,c) bf16
  a2_kernel<<<dim3(CH / 64, NT / 64, NB), dim3(256), 0, stream>>>(
      qkvt, inq, mask, sknow, smask, a2);

  // 3) output projection: out[b][t][j] = A2[b][t][:]·Wout[:][j] + bout[j]
  //    M = NT, N = CH (bias on N), K = CH; f32 output straight into d_out.
  gemm_bf16_wmma<float, false>
      <<<dim3(CH / 128, NT / 64, NB), dim3(256), 0, stream>>>(
          a2,     (size_t)NT * CH,
          woutbf, (size_t)0,
          bout,
          out,    (size_t)NT * CH,
          CH, CH);
}
